// BiLSTM_Parser_simple_62139586839420
// MI455X (gfx1250) — matrix operand order repaired and checked by reading, not compiled, not run on hardware
//
#include <hip/hip_runtime.h>
#include <stddef.h>

// ---------------- Types ----------------
typedef __attribute__((ext_vector_type(16))) _Float16 v16h;
typedef __attribute__((ext_vector_type(8)))  _Float16 h8;
typedef __attribute__((ext_vector_type(8)))  float    v8f;

// Problem constants
#define LSEQ   1024
#define HID    1024
#define HDIR   512
#define G4     2048      // 4*HDIR
#define DIN0   400
#define DIN0P  416       // padded to multiple of 32
#define MLPH   512       // MLP/2
#define NWG    16        // workgroups per direction in the scan

// ---------------- Fragment loaders (ISA 7.12.2 16-bit layouts) ----------------
// A (16x32, MxK): lane<16 -> m=lane, K = kb+{0..7} U kb+{16..23}
//                 lane>=16 -> m=lane-16, K = kb+{8..15} U kb+{24..31}
__device__ __forceinline__ v16h load_a_frag(const _Float16* __restrict__ A,
                                            int K, int m, int kb, bool hi) {
  union { uint4 u[2]; v16h v; } r;
  const _Float16* p = A + (size_t)m * K + kb + (hi ? 8 : 0);
  r.u[0] = *(const uint4*)(p);        // 8 halves
  r.u[1] = *(const uint4*)(p + 16);   // +16 elements
  return r.v;
}
// B (32x16, KxN) with B stored as W[N][K] row-major (B(k,n)=W[n][k]):
// lane<16 -> n=lane, K = kb+{0..7} U kb+{16..23} ; lane>=16 -> n=lane-16, K = kb+{8..15} U kb+{24..31}
__device__ __forceinline__ v16h load_b_frag(const _Float16* __restrict__ W,
                                            int K, int n, int kb, bool hi) {
  union { uint4 u[2]; v16h v; } r;
  const _Float16* p = W + (size_t)n * K + kb + (hi ? 8 : 0);
  r.u[0] = *(const uint4*)(p);
  r.u[1] = *(const uint4*)(p + 16);
  return r.v;
}

// ---------------- WMMA GEMM: C[M,N] = A[M,K] * W[N,K]^T + bias[N] ----------------
// block = 256 threads (8 waves); block tile 128(M) x 64(N); wave tile 32x32.
__global__ __launch_bounds__(256) void gemm_f16_wmma(
    const _Float16* __restrict__ A, const _Float16* __restrict__ W,
    const float* __restrict__ bias, float* __restrict__ C,
    int M, int N, int K) {
  const int lane  = threadIdx.x & 31;
  const int wave  = threadIdx.x >> 5;
  const bool hi   = lane >= 16;
  const int  l15  = lane & 15;
  const int tm = blockIdx.y * 128 + (wave >> 1) * 32;
  const int tn = blockIdx.x * 64  + (wave & 1)  * 32;

  const float bv0 = bias[tn + l15];
  const float bv1 = bias[tn + 16 + l15];
  v8f c00, c01, c10, c11;
#pragma unroll
  for (int r = 0; r < 8; ++r) { c00[r] = bv0; c01[r] = bv1; c10[r] = bv0; c11[r] = bv1; }

  const int m0 = tm + l15, m1 = tm + 16 + l15;
  const int n0 = tn + l15, n1 = tn + 16 + l15;

  for (int kb = 0; kb < K; kb += 32) {
    v16h a0 = load_a_frag(A, K, m0, kb, hi);
    v16h a1 = load_a_frag(A, K, m1, kb, hi);
    v16h b0 = load_b_frag(W, K, n0, kb, hi);
    v16h b1 = load_b_frag(W, K, n1, kb, hi);
    c00 = __builtin_amdgcn_wmma_f32_16x16x32_f16(false, a0, false, b0, (short)0, c00, false, false);
    c01 = __builtin_amdgcn_wmma_f32_16x16x32_f16(false, a0, false, b1, (short)0, c01, false, false);
    c10 = __builtin_amdgcn_wmma_f32_16x16x32_f16(false, a1, false, b0, (short)0, c10, false, false);
    c11 = __builtin_amdgcn_wmma_f32_16x16x32_f16(false, a1, false, b1, (short)0, c11, false, false);
  }
  // C/D layout: VGPR r -> m = tile_m + r + (hi?8:0), n = tile_n + (lane&15)
#pragma unroll
  for (int r = 0; r < 8; ++r) {
    const int row0 = tm + r + (hi ? 8 : 0);
    const int row1 = row0 + 16;
    C[(size_t)row0 * N + tn + l15]      = c00[r];
    C[(size_t)row0 * N + tn + 16 + l15] = c01[r];
    C[(size_t)row1 * N + tn + l15]      = c10[r];
    C[(size_t)row1 * N + tn + 16 + l15] = c11[r];
  }
}

// ---------------- Helpers ----------------
__global__ void zero_u32(unsigned int* p, int n) {
  int i = blockIdx.x * blockDim.x + threadIdx.x;
  if (i < n) p[i] = 0u;
}

// f32 -> f16 row copy with K padding (pad with zeros)
__global__ void cvt_f16(const float* __restrict__ src, _Float16* __restrict__ dst,
                        int sk, int dk) {
  const int row = blockIdx.x;
  for (int k = threadIdx.x; k < dk; k += blockDim.x) {
    float v = (k < sk) ? src[(size_t)row * sk + k] : 0.f;
    dst[(size_t)row * dk + k] = (_Float16)v;
  }
}

// embedding gather: emb16[t, 0:300]=word_emb[wid], [300:400]=pos_emb[pid], [400:416]=0
__global__ void gather_emb(const int* __restrict__ wt, const int* __restrict__ pt,
                           const float* __restrict__ we, const float* __restrict__ pe,
                           _Float16* __restrict__ dst) {
  const int t = blockIdx.x;
  const int wid = wt[t], pid = pt[t];
  for (int k = threadIdx.x; k < DIN0P; k += blockDim.x) {
    float v = 0.f;
    if (k < 300)      v = we[(size_t)wid * 300 + k];
    else if (k < 400) v = pe[(size_t)pid * 100 + (k - 300)];
    dst[(size_t)t * DIN0P + k] = (_Float16)v;
  }
}

// ---------------- Cooperative bidirectional LSTM scan ----------------
// grid = 32 blocks: 0..15 forward, 16..31 backward. block = 256 threads.
// Each block owns 32 hidden units (= 128 gate rows; 2 threads per row).
// Whh slice lives in VGPRs for the whole scan (256 f16/thread = 128 VGPRs);
// per-step traffic is only h (2KB) + xp (512B) per WG. Per-step sync via
// atomic counters; h double-buffered in global (race-free given the barrier).
__global__ __launch_bounds__(256) void lstm_scan(
    const float*    __restrict__ xp,    // [2][L][2048] precomputed input proj
    const _Float16* __restrict__ whh,   // [2][2048][512] f16
    _Float16*       __restrict__ hs16,  // [L][1024] f16 output (dir*512 offset)
    float*          __restrict__ hbuf,  // [2 buf][2 dir][512]
    unsigned int*   __restrict__ cnt)   // [2 dir][L]
{
  __shared__ __align__(16) float h_sh[HDIR];
  __shared__ float g_sh[128];
  const int tid  = threadIdx.x;
  const int dir  = blockIdx.x >> 4;
  const int wg   = blockIdx.x & 15;
  const int u0   = wg * 32;
  const int r    = tid >> 1;          // local gate row 0..127
  const int half = tid & 1;           // which half of the K=512 dot
  const int grow = (r >> 5) * HDIR + u0 + (r & 31);  // global gate row
  const _Float16* wrow = whh + ((size_t)dir * G4 + grow) * HDIR + half * 256;

  // preload this thread's 256-wide weight slice into registers (32 x b128)
  h8 wv[32];
#pragma unroll
  for (int i = 0; i < 32; ++i) wv[i] = *(const h8*)(wrow + i * 8);

  float c = 0.f;  // cell state lives in registers of threads 0..31

  for (int tau = 0; tau < LSEQ; ++tau) {
    const int t = dir ? (LSEQ - 1 - tau) : tau;
    // stage h_{prev} into LDS
    if (tau == 0) {
      h_sh[tid] = 0.f; h_sh[tid + 256] = 0.f;
    } else {
      const float* hp = hbuf + ((size_t)(((tau & 1) ^ 1) * 2 + dir)) * HDIR;
      h_sh[tid] = hp[tid]; h_sh[tid + 256] = hp[tid + 256];
    }
    __syncthreads();

    // half-dot of Whh[grow, :] . h   (weights in VGPRs, h via ds_load_b128)
    float acc = 0.f;
    const float* hh = h_sh + half * 256;
#pragma unroll
    for (int i = 0; i < 32; ++i) {
      const float4 ha = *(const float4*)(hh + i * 8);
      const float4 hb = *(const float4*)(hh + i * 8 + 4);
      acc += (float)wv[i][0] * ha.x + (float)wv[i][1] * ha.y
           + (float)wv[i][2] * ha.z + (float)wv[i][3] * ha.w
           + (float)wv[i][4] * hb.x + (float)wv[i][5] * hb.y
           + (float)wv[i][6] * hb.z + (float)wv[i][7] * hb.w;
    }
    acc += __shfl_xor(acc, 1, 32);
    if (half == 0)
      g_sh[r] = acc + xp[((size_t)dir * LSEQ + t) * G4 + grow];
    __syncthreads();

    if (tid < 32) {
      const int u = tid;
      const float gi = g_sh[u], gf = g_sh[32 + u], gg = g_sh[64 + u], go = g_sh[96 + u];
      const float i_ = 1.f / (1.f + expf(-gi));
      const float f_ = 1.f / (1.f + expf(-gf));
      const float g_ = tanhf(gg);
      const float o_ = 1.f / (1.f + expf(-go));
      c = f_ * c + i_ * g_;
      const float h = o_ * tanhf(c);
      hbuf[((size_t)((tau & 1) * 2 + dir)) * HDIR + u0 + u] = h;
      hs16[(size_t)t * HID + dir * HDIR + u0 + u] = (_Float16)h;
      __threadfence();   // release h before signaling
    }
    __syncthreads();

    if (tid == 0) {
      const unsigned idx = dir * LSEQ + tau;
      atomicAdd(&cnt[idx], 1u);
      while (__atomic_load_n(&cnt[idx], __ATOMIC_ACQUIRE) < NWG)
        __builtin_amdgcn_s_sleep(1);
    }
    __syncthreads();
  }
}

// ---------------- tanh(feat) . wo reductions ----------------
__global__ __launch_bounds__(256) void tanh_dot(
    const float* __restrict__ hf, const float* __restrict__ mf,
    const float* __restrict__ wo, float* __restrict__ sh, float* __restrict__ sm) {
  __shared__ float rh[256], rm[256];
  const int t = blockIdx.x, tid = threadIdx.x;
  float ah = 0.f, am = 0.f;
  for (int j = tid; j < MLPH; j += 256) {
    ah += tanhf(hf[(size_t)t * MLPH + j]) * wo[j];
    am += tanhf(mf[(size_t)t * MLPH + j]) * wo[MLPH + j];
  }
  rh[tid] = ah; rm[tid] = am;
  __syncthreads();
  for (int s = 128; s > 0; s >>= 1) {
    if (tid < s) { rh[tid] += rh[tid + s]; rm[tid] += rm[tid + s]; }
    __syncthreads();
  }
  if (tid == 0) { sh[t] = rh[0]; sm[t] = rm[0]; }
}

// ---------------- final rank-1 score matrix ----------------
__global__ void score_kernel(const float* __restrict__ sh, const float* __restrict__ sm,
                             const float* __restrict__ bo, float* __restrict__ out) {
  const int idx = blockIdx.x * blockDim.x + threadIdx.x;
  const int m = idx >> 10, h = idx & 1023;
  out[idx] = sm[m] + sh[h] + bo[0];
}

// ---------------- host-side orchestration ----------------
extern "C" void kernel_launch(void* const* d_in, const int* in_sizes, int n_in,
                              void* d_out, int out_size, void* d_ws, size_t ws_size,
                              hipStream_t stream) {
  const int*   word = (const int*)  d_in[0];
  const int*   pos  = (const int*)  d_in[1];
  const float* wemb = (const float*)d_in[2];
  const float* pemb = (const float*)d_in[3];
  const float* Wih0 = (const float*)d_in[4];
  const float* Whh0 = (const float*)d_in[5];
  const float* b0   = (const float*)d_in[6];
  const float* Wih1 = (const float*)d_in[7];
  const float* Whh1 = (const float*)d_in[8];
  const float* b1   = (const float*)d_in[9];
  const float* Wh   = (const float*)d_in[10];
  const float* bh   = (const float*)d_in[11];
  const float* Wm   = (const float*)d_in[12];
  const float* bm   = (const float*)d_in[13];
  const float* wo   = (const float*)d_in[14];
  const float* bo   = (const float*)d_in[15];
  float* out = (float*)d_out;

  // workspace carve-up (256B aligned)
  char* base = (char*)d_ws;
  size_t off = 0;
  auto alloc = [&](size_t bytes) -> char* {
    char* p = base + off;
    off = (off + bytes + 255) & ~(size_t)255;
    return p;
  };
  _Float16* emb16   = (_Float16*)alloc((size_t)LSEQ * DIN0P * 2);
  _Float16* wih0_16 = (_Float16*)alloc((size_t)2 * G4 * DIN0P * 2);
  _Float16* whh0_16 = (_Float16*)alloc((size_t)2 * G4 * HDIR * 2);
  _Float16* wih1_16 = (_Float16*)alloc((size_t)2 * G4 * HID * 2);
  _Float16* whh1_16 = (_Float16*)alloc((size_t)2 * G4 * HDIR * 2);
  _Float16* wh16    = (_Float16*)alloc((size_t)MLPH * HID * 2);
  _Float16* wm16    = (_Float16*)alloc((size_t)MLPH * HID * 2);
  float* xp0 = (float*)alloc((size_t)2 * LSEQ * G4 * 4);
  float* xp1 = (float*)alloc((size_t)2 * LSEQ * G4 * 4);
  _Float16* h0_16 = (_Float16*)alloc((size_t)LSEQ * HID * 2);
  _Float16* h1_16 = (_Float16*)alloc((size_t)LSEQ * HID * 2);
  float* hf = (float*)alloc((size_t)LSEQ * MLPH * 4);
  float* mf = (float*)alloc((size_t)LSEQ * MLPH * 4);
  float* s_head = (float*)alloc(LSEQ * 4);
  float* s_mod  = (float*)alloc(LSEQ * 4);
  float* hbuf   = (float*)alloc((size_t)2 * 2 * HDIR * 4);
  unsigned int* cnt = (unsigned int*)alloc((size_t)2 * 2 * LSEQ * 4);  // [layer][dir][L]

  // reset per-step barrier counters every call (graph replays reuse buffers)
  {
    const int n = 2 * 2 * LSEQ;
    zero_u32<<<(n + 255) / 256, 256, 0, stream>>>(cnt, n);
  }

  // stage inputs -> f16
  gather_emb<<<LSEQ, 128, 0, stream>>>(word, pos, wemb, pemb, emb16);
  cvt_f16<<<2 * G4, 256, 0, stream>>>(Wih0, wih0_16, DIN0, DIN0P);
  cvt_f16<<<2 * G4, 256, 0, stream>>>(Whh0, whh0_16, HDIR, HDIR);
  cvt_f16<<<2 * G4, 256, 0, stream>>>(Wih1, wih1_16, HID, HID);
  cvt_f16<<<2 * G4, 256, 0, stream>>>(Whh1, whh1_16, HDIR, HDIR);
  cvt_f16<<<MLPH, 256, 0, stream>>>(Wh, wh16, HID, HID);
  cvt_f16<<<MLPH, 256, 0, stream>>>(Wm, wm16, HID, HID);

  // layer 0 input projections (WMMA): xp0[d] = emb @ Wih0[d]^T + b0[d]
  {
    dim3 grid(G4 / 64, LSEQ / 128);
    for (int d = 0; d < 2; ++d)
      gemm_f16_wmma<<<grid, 256, 0, stream>>>(
          emb16, wih0_16 + (size_t)d * G4 * DIN0P, b0 + (size_t)d * G4,
          xp0 + (size_t)d * LSEQ * G4, LSEQ, G4, DIN0P);
  }
  // layer 0 recurrence (writes f16 hidden states directly)
  lstm_scan<<<32, 256, 0, stream>>>(xp0, whh0_16, h0_16, hbuf, cnt);

  // layer 1 input projections
  {
    dim3 grid(G4 / 64, LSEQ / 128);
    for (int d = 0; d < 2; ++d)
      gemm_f16_wmma<<<grid, 256, 0, stream>>>(
          h0_16, wih1_16 + (size_t)d * G4 * HID, b1 + (size_t)d * G4,
          xp1 + (size_t)d * LSEQ * G4, LSEQ, G4, HID);
  }
  // layer 1 recurrence
  lstm_scan<<<32, 256, 0, stream>>>(xp1, whh1_16, h1_16, hbuf, cnt + 2 * LSEQ);

  // MLP head feature GEMMs (WMMA)
  {
    dim3 grid(MLPH / 64, LSEQ / 128);
    gemm_f16_wmma<<<grid, 256, 0, stream>>>(h1_16, wh16, bh, hf, LSEQ, MLPH, HID);
    gemm_f16_wmma<<<grid, 256, 0, stream>>>(h1_16, wm16, bm, mf, LSEQ, MLPH, HID);
  }

  // tanh + wo dot, then rank-1 broadcast score matrix
  tanh_dot<<<LSEQ, 256, 0, stream>>>(hf, mf, wo, s_head, s_mod);
  score_kernel<<<(LSEQ * LSEQ) / 256, 256, 0, stream>>>(s_head, s_mod, bo, out);

  (void)in_sizes; (void)n_in; (void)out_size; (void)ws_size;
}
